// Attention_14147622273640
// MI455X (gfx1250) — matrix operand order repaired and checked by reading, not compiled, not run on hardware
//
#include <hip/hip_runtime.h>
#include <hip/hip_bf16.h>

// ---------------------------------------------------------------------------
// MLA attention pipeline for MI455X (gfx1250, wave32, WMMA bf16).
// Heavy GEMMs -> v_wmma_f32_16x16x32_bf16 (fp32 accum).
// GEMM LDS staging -> double-buffered global_load_async_to_lds_b128:
// next K-tile DMA overlaps current K-tile WMMA (ASYNCcnt completes in order,
// so s_wait_asynccnt 8 releases the current buffer while the next is in
// flight).
// ---------------------------------------------------------------------------

typedef unsigned short u16;
typedef __attribute__((ext_vector_type(16))) __bf16 v16bf;
typedef __attribute__((ext_vector_type(8)))  float  v8f;

#define B_    2
#define T_    2048
#define C_    2048
#define NH_   16
#define HS_   128
#define NLQ_  512
#define NLKV_ 512
#define DHR_  64

struct alignas(16) U128 { unsigned int x[4]; };

__device__ __forceinline__ u16 f2bf(float f) {
  unsigned int u = __float_as_uint(f);
  u = (u + 0x7FFFu + ((u >> 16) & 1u)) >> 16;   // round-to-nearest-even
  return (u16)u;
}
__device__ __forceinline__ float bf2f(u16 u) {
  return __uint_as_float(((unsigned int)u) << 16);
}

// Async DMA: 16B per lane, global -> LDS, tracked by ASYNCcnt.
__device__ __forceinline__ void async_ld_b128(unsigned lds_off, const void* g) {
  asm volatile("global_load_async_to_lds_b128 %0, %1, off"
               :: "v"(lds_off), "v"(g) : "memory");
}
#define WAIT_ASYNC_0() asm volatile("s_wait_asynccnt 0x0" ::: "memory")
#define WAIT_ASYNC_8() asm volatile("s_wait_asynccnt 0x8" ::: "memory")

// A-fragment (16xK tile, row-major K-contiguous, 16-bit):
// lane m (m<16): K[0..7],[16..23] of row m; lane m+16: K[8..15],[24..31].
__device__ __forceinline__ v16bf fragA16(const u16* base, int ld) {
  int lane = threadIdx.x & 31;
  const u16* p = base + (size_t)(lane & 15) * ld + ((lane >> 4) << 3);
  union { v16bf v; U128 q[2]; } f;
  f.q[0] = *(const U128*)p;
  f.q[1] = *(const U128*)(p + 16);
  return f.v;
}
// B-fragment from N-major rows (K-contiguous): lane n: K[0..7],[16..23]; lane n+16: K[8..15],[24..31].
__device__ __forceinline__ v16bf fragB16(const u16* base, int ld) {
  int lane = threadIdx.x & 31;
  const u16* p = base + (size_t)(lane & 15) * ld + ((lane >> 4) << 3);
  union { v16bf v; U128 q[2]; } f;
  f.q[0] = *(const U128*)p;
  f.q[1] = *(const U128*)(p + 16);
  return f.v;
}
__device__ __forceinline__ v8f wmma_bf16(v16bf a, v16bf b, v8f c) {
  return __builtin_amdgcn_wmma_f32_16x16x32_bf16(false, a, false, b,
                                                 (short)0, c, false, false);
}

// ---------------------------------------------------------------------------
// Prep kernels (fp32 -> bf16 + transposes so every GEMM is A*B^T, K-contig).
// ---------------------------------------------------------------------------
__global__ void cvt_bf16_kernel(u16* __restrict__ dst, const float* __restrict__ src, int n) {
  int i = blockIdx.x * blockDim.x + threadIdx.x;
  if (i < n) dst[i] = f2bf(src[i]);
}
// UkT[h][k][s] = W_uk[(h*128+s)*512 + k]
__global__ void pack_uk_kernel(u16* __restrict__ dst, const float* __restrict__ W_uk) {
  int i = blockIdx.x * blockDim.x + threadIdx.x;
  if (i >= NH_ * NLKV_ * HS_) return;
  int s = i & 127, k = (i >> 7) & 511, h = i >> 16;
  dst[i] = f2bf(W_uk[(size_t)(h * HS_ + s) * NLKV_ + k]);
}
// Uq[h][q][s] = W_uq_flat[q*2048 + h*128 + s]
__global__ void pack_uq_kernel(u16* __restrict__ dst, const float* __restrict__ W_uq) {
  int i = blockIdx.x * blockDim.x + threadIdx.x;
  if (i >= NH_ * NLQ_ * HS_) return;
  int s = i & 127, q = (i >> 7) & 511, h = i >> 16;
  dst[i] = f2bf(W_uq[(size_t)q * (NH_ * HS_) + h * HS_ + s]);
}
// WuvT[k][c] = W_uv[c*512 + k]
__global__ void pack_uvT_kernel(u16* __restrict__ dst, const float* __restrict__ W_uv) {
  int i = blockIdx.x * blockDim.x + threadIdx.x;
  if (i >= NLKV_ * C_) return;
  int c = i & 2047, k = i >> 11;
  dst[i] = f2bf(W_uv[(size_t)c * NLKV_ + k]);
}
// In-place rope on bf16 rows of (B*T, heads*64).
__global__ void rope_kernel(u16* __restrict__ data, const float* __restrict__ cosT,
                            const float* __restrict__ sinT, int heads, int n) {
  int i = blockIdx.x * blockDim.x + threadIdx.x;
  if (i >= n) return;
  int p = i & 31;
  int rem = i >> 5;
  int h = rem % heads;
  int bt = rem / heads;
  int t = bt & (T_ - 1);
  float c = cosT[t * 32 + p], s = sinT[t * 32 + p];
  size_t base = (size_t)bt * heads * DHR_ + h * DHR_ + 2 * p;
  float re = bf2f(data[base]), im = bf2f(data[base + 1]);
  data[base]     = f2bf(re * c - im * s);
  data[base + 1] = f2bf(re * s + im * c);
}

// ---------------------------------------------------------------------------
// Generic batched bf16 WMMA GEMM: Out = A(M x K) * B(N x K)^T.
// 64x64 tile, K-step 64 (8 WMMA per stage), double-buffered async staging.
// Requires M%64==0, N%64==0, K%64==0 (true for all launches here).
// ---------------------------------------------------------------------------
__device__ __forceinline__ void stage_tile(unsigned asB, unsigned bsB,
                                           const u16* Ab, const u16* Bb,
                                           int m0g, int n0g, int lda, int ldb,
                                           int k0, int tid) {
#pragma unroll
  for (int r = 0; r < 4; ++r) {
    int e = r * 1024 + tid * 8;
    int row = e >> 6, col = e & 63;
    unsigned loff = (unsigned)(row * 72 + col) * 2u;
    async_ld_b128(asB + loff, Ab + (size_t)(m0g + row) * lda + k0 + col);
    async_ld_b128(bsB + loff, Bb + (size_t)(n0g + row) * ldb + k0 + col);
  }
}

__global__ __launch_bounds__(128) void gemm_bf16(
    const u16* __restrict__ A, const u16* __restrict__ B, void* __restrict__ O,
    int M, int N, int K, int lda, int ldb, int ldo,
    long long sAi, long long sAj, long long sBi, long long sBj,
    long long sOi, long long sOj, int bdiv, int outBf16) {
  __shared__ __align__(16) u16 As[2][64 * 72];
  __shared__ __align__(16) u16 Bs[2][64 * 72];

  int z = blockIdx.z;
  int bi = z / bdiv, bj = z % bdiv;
  const u16* Ab = A + (size_t)(bi * sAi + bj * sAj);
  const u16* Bb = B + (size_t)(bi * sBi + bj * sBj);
  long long Ooff = bi * sOi + bj * sOj;

  int tid  = threadIdx.x;
  int wave = tid >> 5;
  int wr = wave >> 1, wc = wave & 1;
  int m0g = blockIdx.x * 64;
  int n0g = blockIdx.y * 64;

  unsigned aB[2] = { (unsigned)(size_t)(void*)&As[0][0],
                     (unsigned)(size_t)(void*)&As[1][0] };
  unsigned bB[2] = { (unsigned)(size_t)(void*)&Bs[0][0],
                     (unsigned)(size_t)(void*)&Bs[1][0] };

  v8f acc[2][2];
  acc[0][0] = (v8f)0.0f; acc[0][1] = (v8f)0.0f;
  acc[1][0] = (v8f)0.0f; acc[1][1] = (v8f)0.0f;

  // Prime buffer 0, then pipeline: issue next tile before waiting on current.
  stage_tile(aB[0], bB[0], Ab, Bb, m0g, n0g, lda, ldb, 0, tid);
  int cur = 0;
  for (int k0 = 0; k0 < K; k0 += 64) {
    if (k0 + 64 < K) {
      stage_tile(aB[cur ^ 1], bB[cur ^ 1], Ab, Bb, m0g, n0g, lda, ldb, k0 + 64, tid);
      WAIT_ASYNC_8();          // current tile done; next tile may stay in flight
    } else {
      WAIT_ASYNC_0();
    }
    __syncthreads();           // all waves' DMA for buf[cur] visible

    const u16* Ac = As[cur];
    const u16* Bc = Bs[cur];
#pragma unroll
    for (int ks = 0; ks < 64; ks += 32) {
      v16bf a0 = fragA16(Ac + (wr * 32 +  0) * 72 + ks, 72);
      v16bf a1 = fragA16(Ac + (wr * 32 + 16) * 72 + ks, 72);
      v16bf b0 = fragB16(Bc + (wc * 32 +  0) * 72 + ks, 72);
      v16bf b1 = fragB16(Bc + (wc * 32 + 16) * 72 + ks, 72);
      acc[0][0] = wmma_bf16(a0, b0, acc[0][0]);
      acc[0][1] = wmma_bf16(a0, b1, acc[0][1]);
      acc[1][0] = wmma_bf16(a1, b0, acc[1][0]);
      acc[1][1] = wmma_bf16(a1, b1, acc[1][1]);
    }
    __syncthreads();           // buf[cur] free for the iteration after next
    cur ^= 1;
  }

  // Epilogue: branch hoisted; stores are simple pointer-stride loops.
  int lane = tid & 31, col = lane & 15, hi = lane >> 4;
  if (outBf16) {
    u16* Ob = (u16*)O + Ooff;
#pragma unroll
    for (int mi = 0; mi < 2; ++mi)
#pragma unroll
      for (int nj = 0; nj < 2; ++nj) {
        u16* p = Ob + (size_t)(m0g + wr * 32 + mi * 16 + hi * 8) * ldo
                    + (n0g + wc * 32 + nj * 16 + col);
#pragma unroll
        for (int e = 0; e < 8; ++e) { *p = f2bf(acc[mi][nj][e]); p += ldo; }
      }
  } else {
    float* Ob = (float*)O + Ooff;
#pragma unroll
    for (int mi = 0; mi < 2; ++mi)
#pragma unroll
      for (int nj = 0; nj < 2; ++nj) {
        float* p = Ob + (size_t)(m0g + wr * 32 + mi * 16 + hi * 8) * ldo
                      + (n0g + wc * 32 + nj * 16 + col);
#pragma unroll
        for (int e = 0; e < 8; ++e) { *p = acc[mi][nj][e]; p += ldo; }
      }
  }
}

// ---------------------------------------------------------------------------
// Flash attention over latent keys. One workgroup = (b, h, 16-query tile).
// 4 waves; per 64-key block: wave w scores its 16 keys (K = 512 + 64 rope),
// cross-wave online softmax via LDS, each wave accumulates 16x128 of
// ctx = P * c_kv. Dynamic LDS ~96KB (WGP allows 320KB).
// ---------------------------------------------------------------------------
__global__ __launch_bounds__(128) void mla_flash(
    const u16* __restrict__ qlat,   // (NH, B*T, 512)
    const u16* __restrict__ qr,     // (B*T, NH*64), rope applied
    const u16* __restrict__ ckv,    // (B*T, 512)
    const u16* __restrict__ kr,     // (B*T, 64), rope applied
    u16* __restrict__ ctx) {        // ((b*NH+h)*T, 512)
  extern __shared__ __align__(16) u16 smem[];
  u16* qS  = smem;                      // 16 x 520
  u16* qrS = smem + 16 * 520;           // 16 x 72
  u16* kvT = qrS + 16 * 72;             // 512 x 72 (feature-major c_kv block)
  u16* pS  = kvT + 512 * 72;            // 16 x 72 (probabilities, bf16)
  float* redm = (float*)(pS + 16 * 72); // 4 x 16
  float* redl = redm + 64;              // 4 x 16
  float* mrun = redl + 64;              // 16
  float* lrun = mrun + 16;              // 16

  int t0 = blockIdx.x * 16;
  int h  = blockIdx.y;
  int b  = blockIdx.z;
  int bT = b * T_;
  int tid = threadIdx.x;
  int w = tid >> 5, lane = tid & 31, col = lane & 15, hi = lane >> 4;

#pragma unroll
  for (int it = 0; it < 8; ++it) {
    int lin = it * 1024 + tid * 8;
    int r = lin >> 9, f = lin & 511;
    *(U128*)(qS + r * 520 + f) =
        *(const U128*)(qlat + ((size_t)h * (B_ * T_) + bT + t0 + r) * NLKV_ + f);
  }
  {
    int lin = tid * 8;
    int r = lin >> 6, f = lin & 63;
    *(U128*)(qrS + r * 72 + f) =
        *(const U128*)(qr + (size_t)(bT + t0 + r) * (NH_ * DHR_) + h * DHR_ + f);
  }
  if (tid < 16) { mrun[tid] = -3.0e38f; lrun[tid] = 0.0f; }

  v8f acc[8];
#pragma unroll
  for (int nt = 0; nt < 8; ++nt) acc[nt] = (v8f)0.0f;

  const float rscale = 0.07216878364870322f;   // 1/sqrt(HS + DHR)
  int nblk = (t0 + 16 + 63) >> 6;

  for (int sb = 0; sb < nblk; ++sb) {
    int s0 = sb << 6;
    __syncthreads();   // kvT/pS/redm reusable; staging visible on iter 0

    // transpose-load c_kv block (64 keys x 512 feats) into kvT[f][key]
#pragma unroll 4
    for (int it = 0; it < 32; ++it) {
      int lin = it * 1024 + tid * 8;
      int key = lin >> 9, f = lin & 511;
      union { U128 q; u16 s[8]; } v;
      v.q = *(const U128*)(ckv + (size_t)(bT + s0 + key) * NLKV_ + f);
#pragma unroll
      for (int j = 0; j < 8; ++j) kvT[(f + j) * 72 + key] = v.s[j];
    }
    if (sb + 1 < nblk) {   // pull next key block toward the caches
      const u16* nx = ckv + (size_t)(bT + s0 + 64) * NLKV_ + (size_t)tid * 256;
      __builtin_prefetch(nx, 0, 3);
      __builtin_prefetch(nx + 64, 0, 3);
    }

    // scores: this wave's 16 keys; contraction 512 (latent) + 64 (rope)
    v8f c = (v8f)0.0f;
    const u16* kbase = ckv + (size_t)(bT + s0 + w * 16) * NLKV_;
#pragma unroll
    for (int kt = 0; kt < 16; ++kt)
      c = wmma_bf16(fragA16(qS + kt * 32, 520), fragB16(kbase + kt * 32, NLKV_), c);
    const u16* rbase = kr + (size_t)(bT + s0 + w * 16) * DHR_;
#pragma unroll
    for (int kt = 0; kt < 2; ++kt)
      c = wmma_bf16(fragA16(qrS + kt * 32, 72), fragB16(rbase + kt * 32, DHR_), c);

    // mask + scale + per-row max (rows live in 16-lane halves)
    float sc[8], rmax[8];
#pragma unroll
    for (int e = 0; e < 8; ++e) {
      int tq = t0 + e + hi * 8;
      int sk = s0 + w * 16 + col;
      float v = c[e] * rscale;
      sc[e] = (sk <= tq) ? v : -3.0e38f;
      float m = sc[e];
      m = fmaxf(m, __shfl_xor(m, 1, 32));
      m = fmaxf(m, __shfl_xor(m, 2, 32));
      m = fmaxf(m, __shfl_xor(m, 4, 32));
      m = fmaxf(m, __shfl_xor(m, 8, 32));
      rmax[e] = m;
    }
    if (col == 0)
#pragma unroll
      for (int e = 0; e < 8; ++e) redm[w * 16 + e + hi * 8] = rmax[e];
    __syncthreads();   // redm ready; kvT stores complete

    float pv[8], al[8], ls[8];
#pragma unroll
    for (int e = 0; e < 8; ++e) {
      int row = e + hi * 8;
      float mo = mrun[row];
      float mn = fmaxf(fmaxf(redm[row], redm[16 + row]),
                       fmaxf(redm[32 + row], redm[48 + row]));
      mn = fmaxf(mn, mo);
      pv[e] = __expf(sc[e] - mn);
      al[e] = __expf(mo - mn);
      float s = pv[e];
      s += __shfl_xor(s, 1, 32);
      s += __shfl_xor(s, 2, 32);
      s += __shfl_xor(s, 4, 32);
      s += __shfl_xor(s, 8, 32);
      ls[e] = s;
    }
    if (col == 0)
#pragma unroll
      for (int e = 0; e < 8; ++e) redl[w * 16 + e + hi * 8] = ls[e];
#pragma unroll
    for (int nt = 0; nt < 8; ++nt)
#pragma unroll
      for (int e = 0; e < 8; ++e) acc[nt][e] *= al[e];
#pragma unroll
    for (int e = 0; e < 8; ++e)
      pS[(e + hi * 8) * 72 + w * 16 + col] = f2bf(pv[e]);
    __syncthreads();   // redl + pS ready

    if (tid < 16) {    // one lane per row updates running stats
      float mo = mrun[tid];
      float mn = fmaxf(fmaxf(redm[tid], redm[16 + tid]),
                       fmaxf(redm[32 + tid], redm[48 + tid]));
      mn = fmaxf(mn, mo);
      lrun[tid] = __expf(mo - mn) * lrun[tid] +
                  redl[tid] + redl[16 + tid] + redl[32 + tid] + redl[48 + tid];
      mrun[tid] = mn;
    }

    // ctx += P(16x64) * c_kv(64 x this wave's 128 features)
#pragma unroll
    for (int nt = 0; nt < 8; ++nt)
#pragma unroll
      for (int kt = 0; kt < 2; ++kt)
        acc[nt] = wmma_bf16(fragA16(pS + kt * 32, 72),
                            fragB16(kvT + (size_t)(w * 128 + nt * 16) * 72 + kt * 32, 72),
                            acc[nt]);
  }
  __syncthreads();   // final lrun visible

  float inv[8];
#pragma unroll
  for (int e = 0; e < 8; ++e) inv[e] = 1.0f / lrun[e + hi * 8];
#pragma unroll
  for (int nt = 0; nt < 8; ++nt)
#pragma unroll
    for (int e = 0; e < 8; ++e) {
      size_t off = ((size_t)(b * NH_ + h) * T_ + t0 + e + hi * 8) * NLKV_ +
                   w * 128 + nt * 16 + col;
      ctx[off] = f2bf(acc[nt][e] * inv[e]);
    }
}

// ---------------------------------------------------------------------------
// Host-side pipeline.
// ---------------------------------------------------------------------------
static const int FLASH_SMEM =
    (16 * 520 + 16 * 72 + 512 * 72 + 16 * 72) * 2 + (64 + 64 + 16 + 16) * 4;

extern "C" void kernel_launch(void* const* d_in, const int* in_sizes, int n_in,
                              void* d_out, int out_size, void* d_ws, size_t ws_size,
                              hipStream_t stream) {
  const float* x     = (const float*)d_in[0];
  const float* cosT  = (const float*)d_in[1];
  const float* sinT  = (const float*)d_in[2];
  const float* W_dq  = (const float*)d_in[3];
  const float* W_uq  = (const float*)d_in[4];
  const float* W_dkv = (const float*)d_in[5];
  const float* W_uk  = (const float*)d_in[6];
  const float* W_uv  = (const float*)d_in[7];
  const float* W_qr  = (const float*)d_in[8];
  const float* W_kr  = (const float*)d_in[9];
  const float* W_o   = (const float*)d_in[10];
  float* y = (float*)d_out;

  char* wp = (char*)d_ws;
  auto alloc = [&](size_t bytes) -> void* {
    void* r = (void*)wp;
    wp += (bytes + 255) & ~(size_t)255;
    return r;
  };
  const size_t BT = (size_t)B_ * T_;
  u16* xb    = (u16*)alloc(BT * C_ * 2);
  u16* Wdqb  = (u16*)alloc((size_t)NLQ_ * C_ * 2);
  u16* Wdkvb = (u16*)alloc((size_t)NLKV_ * C_ * 2);
  u16* Wqrb  = (u16*)alloc((size_t)NH_ * DHR_ * NLQ_ * 2);
  u16* Wkrb  = (u16*)alloc((size_t)DHR_ * C_ * 2);
  u16* Wob   = (u16*)alloc((size_t)C_ * C_ * 2);
  u16* UkT   = (u16*)alloc((size_t)NH_ * NLKV_ * HS_ * 2);
  u16* Uq    = (u16*)alloc((size_t)NH_ * NLQ_ * HS_ * 2);
  u16* WuvT  = (u16*)alloc((size_t)NLKV_ * C_ * 2);
  u16* cqb   = (u16*)alloc(BT * NLQ_ * 2);
  u16* ckvb  = (u16*)alloc(BT * NLKV_ * 2);
  u16* keffT = (u16*)alloc((size_t)NH_ * NLKV_ * NLQ_ * 2);
  u16* qlat  = (u16*)alloc((size_t)NH_ * BT * NLKV_ * 2);
  u16* veff  = (u16*)alloc((size_t)C_ * NLKV_ * 2);
  u16* qrp   = (u16*)alloc(BT * NH_ * DHR_ * 2);
  u16* krp   = (u16*)alloc(BT * DHR_ * 2);
  u16* ctxb  = (u16*)alloc((size_t)B_ * NH_ * T_ * NLKV_ * 2);

  auto cvt = [&](u16* dst, const float* src, int n) {
    cvt_bf16_kernel<<<(n + 255) / 256, 256, 0, stream>>>(dst, src, n);
  };
  cvt(xb, x, (int)(BT * C_));
  cvt(Wdqb, W_dq, NLQ_ * C_);
  cvt(Wdkvb, W_dkv, NLKV_ * C_);
  cvt(Wqrb, W_qr, NH_ * DHR_ * NLQ_);
  cvt(Wkrb, W_kr, DHR_ * C_);
  cvt(Wob, W_o, C_ * C_);
  pack_uk_kernel <<<(NH_ * NLKV_ * HS_ + 255) / 256, 256, 0, stream>>>(UkT, W_uk);
  pack_uq_kernel <<<(NH_ * NLQ_ * HS_ + 255) / 256, 256, 0, stream>>>(Uq, W_uq);
  pack_uvT_kernel<<<(NLKV_ * C_ + 255) / 256, 256, 0, stream>>>(WuvT, W_uv);

  auto gemm = [&](const u16* A, const u16* Bm, void* O, int M, int N, int K,
                  int lda, int ldb, int ldo,
                  long long sAi, long long sAj, long long sBi, long long sBj,
                  long long sOi, long long sOj, int nb, int bdiv, int outBf) {
    dim3 g(M / 64, N / 64, nb);
    gemm_bf16<<<g, 128, 0, stream>>>(A, Bm, O, M, N, K, lda, ldb, ldo,
                                     sAi, sAj, sBi, sBj, sOi, sOj, bdiv, outBf);
  };

  // c_q = x * W_dq^T
  gemm(xb, Wdqb, cqb, (int)BT, NLQ_, C_, C_, C_, NLQ_, 0,0,0,0,0,0, 1, 1, 1);
  // c_kv = x * W_dkv^T
  gemm(xb, Wdkvb, ckvb, (int)BT, NLKV_, C_, C_, C_, NLKV_, 0,0,0,0,0,0, 1, 1, 1);
  // k_eff^T[h] = UkT[h](512x128) * Uq[h](512x128)^T
  gemm(UkT, Uq, keffT, NLKV_, NLQ_, HS_, HS_, HS_, NLQ_,
       0, (long long)NLKV_ * HS_, 0, (long long)NLQ_ * HS_,
       0, (long long)NLKV_ * NLQ_, NH_, NH_, 1);
  // q_lat[h] = c_q * k_eff^T[h]^T  (absorbed-weight path)
  gemm(cqb, keffT, qlat, (int)BT, NLKV_, NLQ_, NLQ_, NLQ_, NLKV_,
       0, 0, 0, (long long)NLKV_ * NLQ_,
       0, (long long)BT * NLKV_, NH_, NH_, 1);
  // v_eff_full = W_o * WuvT^T -> (2048,512)
  gemm(Wob, WuvT, veff, C_, NLKV_, C_, C_, C_, NLKV_, 0,0,0,0,0,0, 1, 1, 1);
  // q_r(pre-rope) = c_q * W_qr^T
  gemm(cqb, Wqrb, qrp, (int)BT, NH_ * DHR_, NLQ_, NLQ_, NLQ_, NH_ * DHR_,
       0,0,0,0,0,0, 1, 1, 1);
  // k_r(pre-rope) = x * W_kr^T
  gemm(xb, Wkrb, krp, (int)BT, DHR_, C_, C_, C_, DHR_, 0,0,0,0,0,0, 1, 1, 1);

  {
    int nq = (int)(BT * NH_ * 32);
    rope_kernel<<<(nq + 255) / 256, 256, 0, stream>>>(qrp, cosT, sinT, NH_, nq);
    int nk = (int)(BT * 32);
    rope_kernel<<<(nk + 255) / 256, 256, 0, stream>>>(krp, cosT, sinT, 1, nk);
  }

  (void)hipFuncSetAttribute((const void*)mla_flash,
                            hipFuncAttributeMaxDynamicSharedMemorySize, FLASH_SMEM);
  mla_flash<<<dim3(T_ / 16, NH_, B_), 128, FLASH_SMEM, stream>>>(qlat, qrp, ckvb, krp, ctxb);

  // y[b, t, h*128 + d] = ctx[b,h] * veff[h*128..]^T, fp32 out
  gemm(ctxb, veff, y, T_, HS_, NLKV_, NLKV_, NLKV_, C_,
       (long long)NH_ * T_ * NLKV_, (long long)T_ * NLKV_,
       0, (long long)HS_ * NLKV_,
       (long long)T_ * C_, (long long)HS_,
       B_ * NH_, NH_, 0);
}
